// base_Model_4372276707916
// MI455X (gfx1250) — compile-verified, matrix operand order checked by reading, not run on hardware
//
#include <hip/hip_runtime.h>
#include <hip/hip_bf16.h>

// ---------------- types ----------------
typedef __attribute__((ext_vector_type(8)))  __bf16 v8bf;
typedef __attribute__((ext_vector_type(16))) __bf16 v16bf;
typedef __attribute__((ext_vector_type(8)))  float  v8f;

// ---------------- model dims ----------------
namespace cfg {
constexpr int Bz  = 512;   // batch
constexpr int CIN = 9;
constexpr int Tin = 512;
constexpr int C1  = 32,  L1 = 513, P1 = 257;   // conv1 out len, pooled1 len
constexpr int C2  = 64,  L2 = 258, P2 = 130;   // conv2 out len, pooled2 len (= FEAT)
constexpr int FOC = 64,  HID = 128, H4 = 512, TT = 130;
constexpr int F   = FOC * P2;       // 8320
constexpr int F2  = F / 2;          // 4160
constexpr int PROJ = 256;
}

// ---------------- small helpers ----------------
static __device__ __forceinline__ float sigmf(float x) {
  return 1.f / (1.f + __expf(-x));
}

__global__ void fill_f32_kernel(float* p, int n) {
  int i = blockIdx.x * blockDim.x + threadIdx.x;
  if (i < n) p[i] = 0.f;
}
__global__ void fill_bf16_kernel(__hip_bfloat16* p, int n) {
  int i = blockIdx.x * blockDim.x + threadIdx.x;
  if (i < n) p[i] = __float2bfloat16(0.f);
}
__global__ void f32_to_bf16_2d(const float* __restrict__ src, int sld,
                               __hip_bfloat16* __restrict__ dst, int dld,
                               int rows, int cols) {
  int i = blockIdx.x * blockDim.x + threadIdx.x;
  if (i >= rows * cols) return;
  int r = i / cols, c = i - r * cols;
  dst[(size_t)r * dld + c] = __float2bfloat16(src[(size_t)r * sld + c]);
}
__global__ void bf16_copy_2d(const __hip_bfloat16* __restrict__ src, int sld,
                             __hip_bfloat16* __restrict__ dst, int dld,
                             int rows, int cols) {
  int i = blockIdx.x * blockDim.x + threadIdx.x;
  if (i >= rows * cols) return;
  int r = i / cols, c = i - r * cols;
  dst[(size_t)r * dld + c] = src[(size_t)r * sld + c];
}

// ---------------- conv front-end ----------------
__global__ void conv1d_kernel(const float* __restrict__ x, const float* __restrict__ w,
                              float* __restrict__ y,
                              int Bn, int Cin, int Lin, int Cout, int Lout, int Kw, int pad) {
  int i = blockIdx.x * blockDim.x + threadIdx.x;
  if (i >= Bn * Cout * Lout) return;
  int t = i % Lout; int tmp = i / Lout; int co = tmp % Cout; int b = tmp / Cout;
  float s = 0.f;
  for (int ci = 0; ci < Cin; ++ci) {
    const float* xp = x + ((size_t)b * Cin + ci) * Lin;
    const float* wp = w + ((size_t)co * Cin + ci) * Kw;
    #pragma unroll
    for (int k = 0; k < 8; ++k) {
      int pos = t - pad + k;
      if (pos >= 0 && pos < Lin) s += xp[pos] * wp[k];
    }
  }
  y[i] = s;
}

// training-mode BatchNorm1d statistics over (B, L) per channel (biased var)
__global__ void bn_stats_c_kernel(const float* __restrict__ x, int Bn, int C, int L,
                                  float* __restrict__ mean, float* __restrict__ rstd) {
  __shared__ float ss[256], sq[256];
  int c = blockIdx.x;
  int n = Bn * L;
  float s = 0.f, q = 0.f;
  for (int i = threadIdx.x; i < n; i += blockDim.x) {
    int b = i / L, l = i - b * L;
    float v = x[((size_t)b * C + c) * L + l];
    s += v; q += v * v;
  }
  ss[threadIdx.x] = s; sq[threadIdx.x] = q;
  __syncthreads();
  for (int off = 128; off > 0; off >>= 1) {
    if ((int)threadIdx.x < off) {
      ss[threadIdx.x] += ss[threadIdx.x + off];
      sq[threadIdx.x] += sq[threadIdx.x + off];
    }
    __syncthreads();
  }
  if (threadIdx.x == 0) {
    float m = ss[0] / (float)n;
    float var = sq[0] / (float)n - m * m;
    mean[c] = m;
    rstd[c] = rsqrtf(var + 1e-5f);
  }
}

// BN -> ReLU -> MaxPool1d(kernel=2, stride=2, pad=1, -inf padding)
__global__ void bn_relu_pool_kernel(const float* __restrict__ x,
                                    const float* __restrict__ mean, const float* __restrict__ rstd,
                                    const float* __restrict__ g, const float* __restrict__ bb,
                                    float* __restrict__ y,
                                    int Bn, int C, int L, int Lout) {
  int i = blockIdx.x * blockDim.x + threadIdx.x;
  if (i >= Bn * C * Lout) return;
  int o = i % Lout; int tmp = i / Lout; int c = tmp % C; int b = tmp / C;
  const float* xp = x + ((size_t)b * C + c) * L;
  float m = mean[c], r = rstd[c], gg = g[c], bv = bb[c];
  float best = -3.402823e38f;
  int i0 = 2 * o - 1, i1 = 2 * o;
  if (i0 >= 0 && i0 < L) best = (xp[i0] - m) * r * gg + bv;
  if (i1 < L) { float v = (xp[i1] - m) * r * gg + bv; best = fmaxf(best, v); }
  y[i] = fmaxf(best, 0.f);   // relu commutes with max
}

// pooled2 [B, C, L] f32 -> hflat [B, L*C] bf16  (hidden = transpose(0,2,1))
__global__ void transpose_bf16_kernel(const float* __restrict__ x, __hip_bfloat16* __restrict__ hf,
                                      int Bn, int C, int L) {
  int i = blockIdx.x * blockDim.x + threadIdx.x;
  if (i >= Bn * C * L) return;
  int t = i % L; int tmp = i / L; int c = tmp % C; int b = tmp / C;
  hf[((size_t)b * L + t) * C + c] = __float2bfloat16(x[i]);
}

// ---------------- generic WMMA GEMM ----------------
// D[m][n] = sum_k A[m][k] * W[n][k]  (+ Cin[m][n]) (+ bias[n])
// A, W bf16; accumulate f32 via v_wmma_f32_16x16x32_bf16.
// One wave computes a (16*MREP)x64 output block: MREP M-tiles x 4 N-tiles,
// reusing each B fragment across MREP WMMAs and each A fragment across 4 WMMAs.
// MREP=4 => 16 WMMA per 16 b128 loads (used for the big head GEMMs);
// MREP=1 => max wave-parallelism (used for the latency-bound LSTM step GEMMs).
// K-loop is split into a prefetching steady-state and a branch-free tail so the
// hot loop has no scalar branches around global_prefetch_b8.
// Requirements: M % (16*MREP) == 0, N % 64 == 0, K % 32 == 0; bf16 ptrs 16B aligned.

template <int MREP, bool PF>
static __device__ __forceinline__
void gemm_kstep(const __bf16* __restrict__ A, size_t lda,
                const __bf16* __restrict__ W, size_t ldw,
                int k, int koff, int l16, int mrow0, int ncol0,
                v8f (&acc)[MREP][4]) {
  v16bf av[MREP];
  #pragma unroll
  for (int m = 0; m < MREP; ++m) {
    const __bf16* pa = A + (size_t)(mrow0 + m * 16) * lda + (k + koff);
    if (PF) __builtin_prefetch(pa + 256, 0, 1);   // global_prefetch_b8
    v8bf a0 = *reinterpret_cast<const v8bf*>(pa);
    v8bf a1 = *reinterpret_cast<const v8bf*>(pa + 16);
    av[m] = __builtin_shufflevector(a0, a1, 0,1,2,3,4,5,6,7,8,9,10,11,12,13,14,15);
  }
  #pragma unroll
  for (int j = 0; j < 4; ++j) {
    const __bf16* pb = W + (size_t)(ncol0 + j * 16) * ldw + (k + koff);
    if (PF) __builtin_prefetch(pb + 256, 0, 1);
    v8bf b0 = *reinterpret_cast<const v8bf*>(pb);
    v8bf b1 = *reinterpret_cast<const v8bf*>(pb + 16);
    v16bf bv = __builtin_shufflevector(b0, b1, 0,1,2,3,4,5,6,7,8,9,10,11,12,13,14,15);
    #pragma unroll
    for (int m = 0; m < MREP; ++m) {
      acc[m][j] = __builtin_amdgcn_wmma_f32_16x16x32_bf16(
          false, av[m], false, bv, (short)0, acc[m][j], false, false);
    }
  }
}

template <int MREP>
__global__ __launch_bounds__(128)
void gemm_wmma_bf16(const __bf16* __restrict__ A, int lda,
                    const __bf16* __restrict__ W, int ldw,
                    const float* __restrict__ Cin, int ldcin,
                    const float* __restrict__ bias,
                    float* __restrict__ Cf, __hip_bfloat16* __restrict__ Cb, int ldc,
                    int M, int N, int K) {
  const int ntiles64 = N >> 6;
  const int mgroups = M / (16 * MREP);
  const int wave = blockIdx.x * (blockDim.x >> 5) + ((int)threadIdx.x >> 5);
  if (wave >= mgroups * ntiles64) return;   // wave-uniform: EXEC stays all-ones
  const int lane = threadIdx.x & 31;
  const int mg = wave / ntiles64;
  const int nt = wave - mg * ntiles64;
  const int l16 = lane & 15;
  const int hi  = lane >> 4;              // 0 or 1
  const int koff = hi * 8;
  const int mrow0 = mg * MREP * 16 + l16;   // A row of this lane for m-tile 0
  const int ncol0 = nt * 64 + l16;          // W row (= C col) of this lane for n-tile 0

  v8f acc[MREP][4];
  #pragma unroll
  for (int m = 0; m < MREP; ++m) {
    const int mbase = (mg * MREP + m) * 16 + hi * 8;  // C/D rows for this lane-half
    #pragma unroll
    for (int j = 0; j < 4; ++j) {
      const int n = nt * 64 + j * 16 + l16;
      const float binit = bias ? bias[n] : 0.f;
      #pragma unroll
      for (int v = 0; v < 8; ++v) {
        float c0 = Cin ? Cin[(size_t)(mbase + v) * ldcin + n] : 0.f;
        acc[m][j][v] = c0 + binit;
      }
    }
  }

  // steady state with unconditional prefetch, then branch-free tail
  const int kpf = (K > 256) ? (K - 256) : 0;
  int k = 0;
  for (; k < kpf; k += 32)
    gemm_kstep<MREP, true >(A, lda, W, ldw, k, koff, l16, mrow0, ncol0, acc);
  for (; k < K; k += 32)
    gemm_kstep<MREP, false>(A, lda, W, ldw, k, koff, l16, mrow0, ncol0, acc);

  #pragma unroll
  for (int m = 0; m < MREP; ++m) {
    const int mbase = (mg * MREP + m) * 16 + hi * 8;
    #pragma unroll
    for (int j = 0; j < 4; ++j) {
      const int n = nt * 64 + j * 16 + l16;
      #pragma unroll
      for (int v = 0; v < 8; ++v) {
        const size_t idx = (size_t)(mbase + v) * ldc + n;
        const float x = acc[m][j][v];
        if (Cf) Cf[idx] = x;
        if (Cb) Cb[idx] = __float2bfloat16(x);
      }
    }
  }
}

// ---------------- LSTM cell (gate order i,f,g,o; bias-free) ----------------
__global__ void lstm_cell_kernel(const float* __restrict__ gates, float* __restrict__ c,
                                 __hip_bfloat16* __restrict__ h1, int ldh1, int Bn) {
  int i = blockIdx.x * blockDim.x + threadIdx.x;
  if (i >= Bn * 128) return;
  int b = i >> 7, j = i & 127;
  const float* gp = gates + (size_t)b * 512;
  float ig = sigmf(gp[j]);
  float fg = sigmf(gp[128 + j]);
  float gg = tanhf(gp[256 + j]);
  float og = sigmf(gp[384 + j]);
  float cn = fg * c[i] + ig * gg;
  c[i] = cn;
  h1[(size_t)b * ldh1 + j] = __float2bfloat16(og * tanhf(cn));
}

// ---------------- feature BatchNorm (over batch dim) ----------------
__global__ void bnf_stats_kernel(const float* __restrict__ z, int rows, int cols,
                                 float* __restrict__ mean, float* __restrict__ rstd) {
  int j = blockIdx.x * blockDim.x + threadIdx.x;
  if (j >= cols) return;
  float s = 0.f, q = 0.f;
  for (int r = 0; r < rows; ++r) {
    float v = z[(size_t)r * cols + j];
    s += v; q += v * v;
  }
  float m = s / (float)rows;
  float var = q / (float)rows - m * m;
  mean[j] = m;
  rstd[j] = rsqrtf(var + 1e-5f);
}

__global__ void bnf_apply_kernel(const float* __restrict__ z,
                                 const float* __restrict__ mean, const float* __restrict__ rstd,
                                 const float* __restrict__ g, const float* __restrict__ bb,
                                 __hip_bfloat16* __restrict__ out, int rows, int cols) {
  int i = blockIdx.x * blockDim.x + threadIdx.x;
  if (i >= rows * cols) return;
  int j = i % cols;
  float v = (z[i] - mean[j]) * rstd[j] * g[j] + bb[j];
  out[i] = __float2bfloat16(fmaxf(v, 0.f));
}

// ---------------- host-side launch helpers ----------------
static inline int gsz(int n) { return (n + 255) / 256; }

static inline void launch_gemm(hipStream_t s, const void* A, int lda, const void* W, int ldw,
                               const float* Cin, int ldcin, const float* bias,
                               float* Cf, __hip_bfloat16* Cb, int ldc,
                               int M, int N, int K, bool wide) {
  if (wide) {
    int waves = (M >> 6) * (N >> 6);          // 64x64 block per wave
    int blocks = (waves + 3) >> 2;
    gemm_wmma_bf16<4><<<blocks, 128, 0, s>>>((const __bf16*)A, lda, (const __bf16*)W, ldw,
                                             Cin, ldcin, bias, Cf, Cb, ldc, M, N, K);
  } else {
    int waves = (M >> 4) * (N >> 6);          // 16x64 block per wave
    int blocks = (waves + 3) >> 2;
    gemm_wmma_bf16<1><<<blocks, 128, 0, s>>>((const __bf16*)A, lda, (const __bf16*)W, ldw,
                                             Cin, ldcin, bias, Cf, Cb, ldc, M, N, K);
  }
}

extern "C" void kernel_launch(void* const* d_in, const int* in_sizes, int n_in,
                              void* d_out, int out_size, void* d_ws, size_t ws_size,
                              hipStream_t stream) {
  using namespace cfg;
  // ----- inputs (setup_inputs order) -----
  const float* x_in    = (const float*)d_in[0];
  const float* conv1_w = (const float*)d_in[1];
  const float* bn1_g   = (const float*)d_in[2];
  const float* bn1_b   = (const float*)d_in[3];
  const float* conv2_w = (const float*)d_in[4];
  const float* bn2_g   = (const float*)d_in[5];
  const float* bn2_b   = (const float*)d_in[6];
  const float* enc_wih0 = (const float*)d_in[7];
  const float* enc_whh0 = (const float*)d_in[8];
  const float* enc_wih1 = (const float*)d_in[9];
  const float* enc_whh1 = (const float*)d_in[10];
  const float* dec_wih0 = (const float*)d_in[11];
  const float* dec_whh0 = (const float*)d_in[12];
  const float* dec_wih1 = (const float*)d_in[13];
  const float* dec_whh1 = (const float*)d_in[14];
  const float* out_w   = (const float*)d_in[15];
  const float* out_b   = (const float*)d_in[16];
  const float* ph_w1   = (const float*)d_in[17];
  const float* ph_b1   = (const float*)d_in[18];
  const float* ph_g    = (const float*)d_in[19];
  const float* ph_bb   = (const float*)d_in[20];
  const float* ph_w2   = (const float*)d_in[21];
  const float* ph_b2   = (const float*)d_in[22];

  // ----- workspace bump allocator (256B aligned) -----
  char* base = (char*)d_ws;
  size_t off = 0;
  auto alloc = [&](size_t bytes) -> void* {
    off = (off + 255) & ~(size_t)255;
    void* p = base + off;
    off += bytes;
    return p;
  };
  float* convbuf = (float*)alloc((size_t)Bz * C2 * L2 * 4);   // reused for conv1 & conv2
  float* pooled1 = (float*)alloc((size_t)Bz * C1 * P1 * 4);
  float* pooled2 = (float*)alloc((size_t)Bz * C2 * P2 * 4);
  __hip_bfloat16* hflat = (__hip_bfloat16*)alloc((size_t)Bz * F * 2);
  __hip_bfloat16* seq0  = (__hip_bfloat16*)alloc((size_t)TT * Bz * HID * 2);
  __hip_bfloat16* h1e   = (__hip_bfloat16*)alloc((size_t)Bz * HID * 2);
  __hip_bfloat16* hzero = (__hip_bfloat16*)alloc((size_t)Bz * HID * 2);
  __hip_bfloat16* hdec  = (__hip_bfloat16*)alloc((size_t)Bz * 256 * 2);
  float* c0 = (float*)alloc((size_t)Bz * HID * 4);
  float* c1 = (float*)alloc((size_t)Bz * HID * 4);
  float* gates = (float*)alloc((size_t)Bz * H4 * 4);
  __hip_bfloat16* oflat = (__hip_bfloat16*)alloc((size_t)Bz * F * 2);
  float* z1f = (float*)alloc((size_t)Bz * F2 * 4);
  __hip_bfloat16* z1bf = (__hip_bfloat16*)alloc((size_t)Bz * F2 * 2);
  float* mean1 = (float*)alloc(C1 * 4);
  float* rstd1 = (float*)alloc(C1 * 4);
  float* mean2 = (float*)alloc(C2 * 4);
  float* rstd2 = (float*)alloc(C2 * 4);
  float* bnf_mean = (float*)alloc(F2 * 4);
  float* bnf_rstd = (float*)alloc(F2 * 4);
  __hip_bfloat16* wih0e = (__hip_bfloat16*)alloc((size_t)H4 * FOC * 2);
  __hip_bfloat16* whh0e = (__hip_bfloat16*)alloc((size_t)H4 * HID * 2);
  __hip_bfloat16* wih1e = (__hip_bfloat16*)alloc((size_t)H4 * HID * 2);
  __hip_bfloat16* whh1e = (__hip_bfloat16*)alloc((size_t)H4 * HID * 2);
  __hip_bfloat16* wih0d = (__hip_bfloat16*)alloc((size_t)H4 * FOC * 2);
  __hip_bfloat16* whh0d = (__hip_bfloat16*)alloc((size_t)H4 * HID * 2);
  __hip_bfloat16* w1cat = (__hip_bfloat16*)alloc((size_t)H4 * 256 * 2);  // [wih1 | whh1]
  __hip_bfloat16* outw  = (__hip_bfloat16*)alloc((size_t)FOC * HID * 2);
  __hip_bfloat16* phw1  = (__hip_bfloat16*)alloc((size_t)F2 * F * 2);
  __hip_bfloat16* phw2  = (__hip_bfloat16*)alloc((size_t)PROJ * F2 * 2);
  (void)ws_size; (void)n_in; (void)in_sizes; (void)out_size;

  // ----- front-end: conv1 -> BN -> ReLU -> pool -> conv2 -> BN -> ReLU -> pool -----
  conv1d_kernel<<<gsz(Bz * C1 * L1), 256, 0, stream>>>(x_in, conv1_w, convbuf,
                                                       Bz, CIN, Tin, C1, L1, 8, 4);
  bn_stats_c_kernel<<<C1, 256, 0, stream>>>(convbuf, Bz, C1, L1, mean1, rstd1);
  bn_relu_pool_kernel<<<gsz(Bz * C1 * P1), 256, 0, stream>>>(convbuf, mean1, rstd1,
                                                             bn1_g, bn1_b, pooled1,
                                                             Bz, C1, L1, P1);
  conv1d_kernel<<<gsz(Bz * C2 * L2), 256, 0, stream>>>(pooled1, conv2_w, convbuf,
                                                       Bz, C1, P1, C2, L2, 8, 4);
  bn_stats_c_kernel<<<C2, 256, 0, stream>>>(convbuf, Bz, C2, L2, mean2, rstd2);
  bn_relu_pool_kernel<<<gsz(Bz * C2 * P2), 256, 0, stream>>>(convbuf, mean2, rstd2,
                                                             bn2_g, bn2_b, pooled2,
                                                             Bz, C2, L2, P2);
  // hidden = transpose -> hflat bf16 [B, T*FOC]
  transpose_bf16_kernel<<<gsz(Bz * C2 * P2), 256, 0, stream>>>(pooled2, hflat, Bz, C2, P2);

  // ----- weight conversions to bf16 (every call; deterministic) -----
  f32_to_bf16_2d<<<gsz(H4 * FOC), 256, 0, stream>>>(enc_wih0, FOC, wih0e, FOC, H4, FOC);
  f32_to_bf16_2d<<<gsz(H4 * HID), 256, 0, stream>>>(enc_whh0, HID, whh0e, HID, H4, HID);
  f32_to_bf16_2d<<<gsz(H4 * HID), 256, 0, stream>>>(enc_wih1, HID, wih1e, HID, H4, HID);
  f32_to_bf16_2d<<<gsz(H4 * HID), 256, 0, stream>>>(enc_whh1, HID, whh1e, HID, H4, HID);
  f32_to_bf16_2d<<<gsz(H4 * FOC), 256, 0, stream>>>(dec_wih0, FOC, wih0d, FOC, H4, FOC);
  f32_to_bf16_2d<<<gsz(H4 * HID), 256, 0, stream>>>(dec_whh0, HID, whh0d, HID, H4, HID);
  f32_to_bf16_2d<<<gsz(H4 * HID), 256, 0, stream>>>(dec_wih1, HID, w1cat, 256, H4, HID);
  f32_to_bf16_2d<<<gsz(H4 * HID), 256, 0, stream>>>(dec_whh1, HID, w1cat + HID, 256, H4, HID);
  f32_to_bf16_2d<<<gsz(FOC * HID), 256, 0, stream>>>(out_w, HID, outw, HID, FOC, HID);
  f32_to_bf16_2d<<<gsz(F2 * F), 256, 0, stream>>>(ph_w1, F, phw1, F, F2, F);
  f32_to_bf16_2d<<<gsz(PROJ * F2), 256, 0, stream>>>(ph_w2, F2, phw2, F2, PROJ, F2);

  // ----- state init -----
  fill_f32_kernel<<<gsz(Bz * HID), 256, 0, stream>>>(c0, Bz * HID);
  fill_f32_kernel<<<gsz(Bz * HID), 256, 0, stream>>>(c1, Bz * HID);
  fill_bf16_kernel<<<gsz(Bz * HID), 256, 0, stream>>>(hzero, Bz * HID);

  // ----- encoder layer 0 (scan over T) -----
  for (int t = 0; t < TT; ++t) {
    launch_gemm(stream, hflat + (size_t)t * FOC, F, wih0e, FOC,
                nullptr, 0, nullptr, gates, nullptr, H4, Bz, H4, FOC, false);
    const void* hprev = (t == 0) ? (const void*)hzero
                                 : (const void*)(seq0 + (size_t)(t - 1) * Bz * HID);
    launch_gemm(stream, hprev, HID, whh0e, HID,
                gates, H4, nullptr, gates, nullptr, H4, Bz, H4, HID, false);
    lstm_cell_kernel<<<gsz(Bz * HID), 256, 0, stream>>>(gates, c0,
                                                        seq0 + (size_t)t * Bz * HID, HID, Bz);
  }
  // ----- encoder layer 1 -----
  for (int t = 0; t < TT; ++t) {
    launch_gemm(stream, seq0 + (size_t)t * Bz * HID, HID, wih1e, HID,
                nullptr, 0, nullptr, gates, nullptr, H4, Bz, H4, HID, false);
    const void* hprev = (t == 0) ? (const void*)hzero : (const void*)h1e;
    launch_gemm(stream, hprev, HID, whh1e, HID,
                gates, H4, nullptr, gates, nullptr, H4, Bz, H4, HID, false);
    lstm_cell_kernel<<<gsz(Bz * HID), 256, 0, stream>>>(gates, c1, h1e, HID, Bz);
  }

  // ----- decoder state = encoder finals; hdec = [h0 | h1] per row -----
  bf16_copy_2d<<<gsz(Bz * HID), 256, 0, stream>>>(seq0 + (size_t)(TT - 1) * Bz * HID, HID,
                                                  hdec, 256, Bz, HID);
  bf16_copy_2d<<<gsz(Bz * HID), 256, 0, stream>>>(h1e, HID, hdec + HID, 256, Bz, HID);
  // c0/c1 already hold encoder final cell state; decoder updates in place.

  // ----- decoder (reversed-time, teacher-forced) -----
  for (int i = 0; i < TT; ++i) {
    const int tr = TT - 1 - i;
    // out_t = h0(pre-update) @ out_w.T + out_b  -> oflat at time tr (bf16, head input)
    launch_gemm(stream, hdec, 256, outw, HID,
                nullptr, 0, out_b, nullptr, oflat + (size_t)tr * FOC, F, Bz, FOC, HID, false);
    // layer 0 gates: x_tr @ wih0.T + h0 @ whh0.T
    launch_gemm(stream, hflat + (size_t)tr * FOC, F, wih0d, FOC,
                nullptr, 0, nullptr, gates, nullptr, H4, Bz, H4, FOC, false);
    launch_gemm(stream, hdec, 256, whh0d, HID,
                gates, H4, nullptr, gates, nullptr, H4, Bz, H4, HID, false);
    lstm_cell_kernel<<<gsz(Bz * HID), 256, 0, stream>>>(gates, c0, hdec, 256, Bz);
    // layer 1 gates fused: [h0_new | h1_prev] @ [wih1 | whh1].T  (K=256)
    launch_gemm(stream, hdec, 256, w1cat, 256,
                nullptr, 0, nullptr, gates, nullptr, H4, Bz, H4, 256, false);
    lstm_cell_kernel<<<gsz(Bz * HID), 256, 0, stream>>>(gates, c1, hdec + HID, 256, Bz);
  }

  // ----- projection heads (register-blocked wide WMMA GEMMs) -----
  auto run_head = [&](const __hip_bfloat16* zf, float* outp) {
    launch_gemm(stream, zf, F, phw1, F, nullptr, 0, ph_b1, z1f, nullptr, F2,
                Bz, F2, F, true);
    bnf_stats_kernel<<<gsz(F2), 256, 0, stream>>>(z1f, Bz, F2, bnf_mean, bnf_rstd);
    bnf_apply_kernel<<<gsz(Bz * F2), 256, 0, stream>>>(z1f, bnf_mean, bnf_rstd,
                                                       ph_g, ph_bb, z1bf, Bz, F2);
    launch_gemm(stream, z1bf, F2, phw2, F2, nullptr, 0, ph_b2, outp, nullptr, PROJ,
                Bz, PROJ, F2, true);
  };
  run_head(hflat, (float*)d_out);
  run_head(oflat, (float*)d_out + (size_t)Bz * PROJ);
}
